// MultiGATBaseConvs_52948356825717
// MI455X (gfx1250) — matrix operand order repaired and checked by reading, not compiled, not run on hardware
//
#include <hip/hip_runtime.h>
#include <hip/hip_bf16.h>
#include <stdint.h>

// ---------------------------------------------------------------------------
// Types for CDNA5 WMMA (wave32): A/B are 16 bf16 per lane, C/D are 8 f32.
// ---------------------------------------------------------------------------
typedef __attribute__((ext_vector_type(16))) __bf16 bf16x16;
typedef __attribute__((ext_vector_type(4)))  __bf16 bf16x4;
typedef __attribute__((ext_vector_type(8)))  float  floatx8;

union Frag {
  bf16x16 v;
  uint4   q[2];
};

// monotonic float <-> uint mapping for atomicMax on floats
__device__ __forceinline__ unsigned ford(float f) {
  unsigned u = __float_as_uint(f);
  return (u & 0x80000000u) ? ~u : (u | 0x80000000u);
}
__device__ __forceinline__ float funord(unsigned m) {
  return __uint_as_float((m & 0x80000000u) ? (m ^ 0x80000000u) : ~m);
}

// ---------------------------------------------------------------------------
// GEMM: C[M,N] = A[M,K] @ B[K,N], f32 in memory, bf16 WMMA compute, f32 accum.
// Block tile 128x128, BK=32, double-buffered LDS, one barrier per K-step.
// 256 threads = 8 waves; wave (wm,wn) in 2x4 grid computes 64x32 = 4x2 WMMA
// tiles. LDS: sA[buf][row][k], sB[buf][n][k] (B transposed during staging) so
// per-lane fragments are contiguous b128 loads matching the ISA 16-bit layout:
//   A 16x32: lanes 0-15 = M rows, K chunks {half*8..+7} and {16+half*8..+7}
//   B 32x16: lanes 0-15 = N cols, same K chunks as A
// M,N multiples of 128, K multiple of 32 (true for all six GEMMs here).
// ---------------------------------------------------------------------------
#define AS 40  // LDS row stride (elements); 80B rows keep 16B alignment
#define BS 40

__global__ void __launch_bounds__(256)
gemm_bf16wmma_kernel(const float* __restrict__ A, const float* __restrict__ B,
                     float* __restrict__ C, int M, int N, int K) {
  __shared__ __bf16 sA[2][128 * AS];
  __shared__ __bf16 sB[2][128 * BS];

  const int tid  = threadIdx.x;
  const int lane = tid & 31;
  const int wave = tid >> 5;
  const int wm   = wave & 1;   // 0..1 -> 64 rows each
  const int wn   = wave >> 1;  // 0..3 -> 32 cols each
  const int m0   = blockIdx.y * 128;
  const int n0   = blockIdx.x * 128;
  const int half = lane >> 4;  // lane group 0/1
  const int l16  = lane & 15;

  // per-thread staging coordinates (1024 float4 per tile, 4 per thread)
  const int arow[4] = {(tid + 0) >> 3, (tid + 256) >> 3, (tid + 512) >> 3,
                       (tid + 768) >> 3};
  const int ac4 = tid & 7;
  const int bkr[4] = {(tid + 0) >> 5, (tid + 256) >> 5, (tid + 512) >> 5,
                      (tid + 768) >> 5};
  const int bc4 = tid & 31;

  const floatx8 fzero = {0.f, 0.f, 0.f, 0.f, 0.f, 0.f, 0.f, 0.f};
  floatx8 acc[4][2];
#pragma unroll
  for (int i = 0; i < 4; ++i)
#pragma unroll
    for (int j = 0; j < 2; ++j) acc[i][j] = fzero;

  float4 ra[4], rb[4];

  // ---- preload tile 0 into registers (all loads issued back-to-back)
#pragma unroll
  for (int i = 0; i < 4; ++i)
    ra[i] = *(const float4*)(A + (size_t)(m0 + arow[i]) * (size_t)K + ac4 * 4);
#pragma unroll
  for (int i = 0; i < 4; ++i)
    rb[i] = *(const float4*)(B + (size_t)bkr[i] * (size_t)N + n0 + bc4 * 4);

  // ---- store tile 0 into LDS buffer 0 (native f32->bf16 RNE casts)
#pragma unroll
  for (int i = 0; i < 4; ++i) {
    bf16x4 h = {(__bf16)ra[i].x, (__bf16)ra[i].y, (__bf16)ra[i].z,
                (__bf16)ra[i].w};
    *(bf16x4*)&sA[0][arow[i] * AS + ac4 * 4] = h;
  }
#pragma unroll
  for (int i = 0; i < 4; ++i) {
    __bf16* bp = &sB[0][(bc4 * 4) * BS + bkr[i]];
    bp[0 * BS] = (__bf16)rb[i].x;
    bp[1 * BS] = (__bf16)rb[i].y;
    bp[2 * BS] = (__bf16)rb[i].z;
    bp[3 * BS] = (__bf16)rb[i].w;
  }

  const int nt = K >> 5;  // K/32 tiles
  for (int t = 0; t < nt; ++t) {
    __syncthreads();  // LDS buffer (t&1) ready for all waves
    const int cur = t & 1;
    const bool more = (t + 1) < nt;

    // ---- issue global loads for tile t+1 early (hide latency under WMMA)
    if (more) {
      const int k1 = (t + 1) << 5;
#pragma unroll
      for (int i = 0; i < 4; ++i)
        ra[i] = *(const float4*)(A + (size_t)(m0 + arow[i]) * (size_t)K + k1 +
                                 ac4 * 4);
#pragma unroll
      for (int i = 0; i < 4; ++i)
        rb[i] = *(const float4*)(B + (size_t)(k1 + bkr[i]) * (size_t)N + n0 +
                                 bc4 * 4);
    }

    // ---- fragments from LDS buffer cur + 8 WMMAs
    Frag fa[4], fb[2];
#pragma unroll
    for (int ti = 0; ti < 4; ++ti) {
      const uint4* p =
          (const uint4*)&sA[cur][(wm * 64 + ti * 16 + l16) * AS + half * 8];
      fa[ti].q[0] = p[0];  // K = half*8 .. half*8+7
      fa[ti].q[1] = p[2];  // K = 16+half*8 .. +7  (+32 bytes)
    }
#pragma unroll
    for (int tj = 0; tj < 2; ++tj) {
      const uint4* p =
          (const uint4*)&sB[cur][(wn * 32 + tj * 16 + l16) * BS + half * 8];
      fb[tj].q[0] = p[0];  // K = half*8 .. half*8+7
      fb[tj].q[1] = p[2];  // K = 16+half*8 .. +7  (+32 bytes)
    }
#pragma unroll
    for (int ti = 0; ti < 4; ++ti)
#pragma unroll
      for (int tj = 0; tj < 2; ++tj)
        acc[ti][tj] = __builtin_amdgcn_wmma_f32_16x16x32_bf16(
            false, fa[ti].v, false, fb[tj].v, (short)0, acc[ti][tj], false,
            false);

    // ---- convert + store tile t+1 into the other buffer (no barrier needed:
    //      the barrier at top of iteration t guaranteed everyone finished
    //      reading buffer cur^1 during iteration t-1)
    if (more) {
      const int nxt = cur ^ 1;
#pragma unroll
      for (int i = 0; i < 4; ++i) {
        bf16x4 h = {(__bf16)ra[i].x, (__bf16)ra[i].y, (__bf16)ra[i].z,
                    (__bf16)ra[i].w};
        *(bf16x4*)&sA[nxt][arow[i] * AS + ac4 * 4] = h;
      }
#pragma unroll
      for (int i = 0; i < 4; ++i) {
        __bf16* bp = &sB[nxt][(bc4 * 4) * BS + bkr[i]];
        bp[0 * BS] = (__bf16)rb[i].x;
        bp[1 * BS] = (__bf16)rb[i].y;
        bp[2 * BS] = (__bf16)rb[i].z;
        bp[3 * BS] = (__bf16)rb[i].w;
      }
    }
  }

  // ---- epilogue: C/D layout = VGPR r -> M = r + 8*half, N = l16
#pragma unroll
  for (int ti = 0; ti < 4; ++ti) {
#pragma unroll
    for (int tj = 0; tj < 2; ++tj) {
      int row = m0 + wm * 64 + ti * 16 + half * 8;
      int col = n0 + wn * 32 + tj * 16 + l16;
      float* cp = C + (size_t)row * (size_t)N + col;
#pragma unroll
      for (int r = 0; r < 8; ++r) cp[(size_t)r * (size_t)N] = acc[ti][tj][r];
    }
  }
}

// ---------------------------------------------------------------------------
// el[n,h] = <ft[n,h,:], al[h,:]>, er likewise. One wave per (n,h) row.
// ---------------------------------------------------------------------------
__global__ void __launch_bounds__(256)
scores_kernel(const float* __restrict__ ft, const float* __restrict__ al,
              const float* __restrict__ ar, float* __restrict__ el,
              float* __restrict__ er, int NH, int H, int D) {
  int gwave = (int)((blockIdx.x * blockDim.x + threadIdx.x) >> 5);
  int lane  = threadIdx.x & 31;
  if (gwave >= NH) return;
  int n = gwave / H, h = gwave - n * H;
  const float* f = ft + (size_t)n * (size_t)(H * D) + (size_t)h * D;
  const float* a = al + (size_t)h * D;
  const float* b = ar + (size_t)h * D;
  float sl = 0.f, sr = 0.f;
  for (int d = lane; d < D; d += 32) {
    float v = f[d];
    sl += v * a[d];
    sr += v * b[d];
  }
#pragma unroll
  for (int off = 16; off > 0; off >>= 1) {
    sl += __shfl_down(sl, off);
    sr += __shfl_down(sr, off);
  }
  if (lane == 0) { el[gwave] = sl; er[gwave] = sr; }
}

__global__ void zero_kernel(float* __restrict__ p, size_t n) {
  size_t i = (size_t)blockIdx.x * blockDim.x + threadIdx.x;
  if (i < n) p[i] = 0.f;
}

__global__ void edge_max_kernel(const float* __restrict__ el,
                                const float* __restrict__ er,
                                const int* __restrict__ src,
                                const int* __restrict__ dst,
                                float* __restrict__ ebuf,
                                unsigned* __restrict__ nmax, int EH, int H) {
  int i = (int)(blockIdx.x * blockDim.x + threadIdx.x);
  if (i >= EH) return;
  int e = i / H, h = i - e * H;
  float v = el[src[e] * H + h] + er[dst[e] * H + h];
  v = v > 0.f ? v : 0.2f * v;  // leaky_relu, slope 0.2
  ebuf[i] = v;
  atomicMax(&nmax[dst[e] * H + h], ford(v));
}

__global__ void edge_exp_kernel(float* __restrict__ ebuf,
                                const unsigned* __restrict__ nmax,
                                float* __restrict__ nsum,
                                const int* __restrict__ dst, int EH, int H) {
  int i = (int)(blockIdx.x * blockDim.x + threadIdx.x);
  if (i >= EH) return;
  int e = i / H, h = i - e * H;
  int d = dst[e];
  float ex = expf(ebuf[i] - funord(nmax[d * H + h]));
  ebuf[i] = ex;
  atomicAdd(&nsum[d * H + h], ex);
}

__global__ void edge_norm_kernel(const float* __restrict__ ebuf,
                                 const float* __restrict__ nsum,
                                 const int* __restrict__ dst,
                                 float* __restrict__ abuf, int EH, int H) {
  int i = (int)(blockIdx.x * blockDim.x + threadIdx.x);
  if (i >= EH) return;
  int e = i / H, h = i - e * H;
  abuf[i] = ebuf[i] / nsum[dst[e] * H + h];
}

// agg[dst,h,d] += ft[src,h,d] * a[e,h]; one block per edge, dshift = log2(D)
__global__ void __launch_bounds__(256)
aggregate_kernel(const float* __restrict__ ft, const float* __restrict__ attn,
                 const int* __restrict__ src, const int* __restrict__ dst,
                 float* __restrict__ agg, int HD, int dshift) {
  int e = blockIdx.x;
  int s = src[e], d = dst[e];
  const float* f = ft + (size_t)s * HD;
  float*       g = agg + (size_t)d * HD;
  const float* a = attn + (size_t)e * (size_t)(HD >> dshift);
  for (int i = threadIdx.x; i < HD; i += blockDim.x)
    atomicAdd(&g[i], f[i] * a[i >> dshift]);
}

__global__ void add_relu_kernel(const float* __restrict__ a,
                                const float* __restrict__ b,
                                float* __restrict__ o, size_t n) {
  size_t i = (size_t)blockIdx.x * blockDim.x + threadIdx.x;
  if (i < n) o[i] = fmaxf(a[i] + b[i], 0.f);
}

// out = [ relu(agg4+res4) (N*D) | attn (E) | agg4 (N*D) ]
__global__ void final_kernel(const float* __restrict__ agg4,
                             const float* __restrict__ res4,
                             const float* __restrict__ attn,
                             float* __restrict__ out, int ND, int E) {
  int i = (int)(blockIdx.x * blockDim.x + threadIdx.x);
  int total = ND + E + ND;
  if (i >= total) return;
  if (i < ND)          out[i] = fmaxf(agg4[i] + res4[i], 0.f);
  else if (i < ND + E) out[i] = attn[i - ND];
  else                 out[i] = agg4[i - ND - E];
}

// ---------------------------------------------------------------------------
extern "C" void kernel_launch(void* const* d_in, const int* in_sizes, int n_in,
                              void* d_out, int out_size, void* d_ws,
                              size_t ws_size, hipStream_t stream) {
  (void)in_sizes; (void)n_in; (void)out_size; (void)ws_size;
  const float* feat  = (const float*)d_in[0];
  const int*   src   = (const int*)d_in[1];
  const int*   dst   = (const int*)d_in[2];
  const float* W1    = (const float*)d_in[3];
  const float* al1   = (const float*)d_in[4];
  const float* ar1   = (const float*)d_in[5];
  const float* resW1 = (const float*)d_in[6];
  const float* W2    = (const float*)d_in[7];
  const float* al2   = (const float*)d_in[8];
  const float* ar2   = (const float*)d_in[9];
  const float* W3    = (const float*)d_in[10];
  const float* al3   = (const float*)d_in[11];
  const float* ar3   = (const float*)d_in[12];
  const float* W4    = (const float*)d_in[13];
  const float* al4   = (const float*)d_in[14];
  const float* ar4   = (const float*)d_in[15];
  const float* resW4 = (const float*)d_in[16];

  const int N = 2048, E = 32768, C = 512, H = 16, D = 512, F = H * D;  // 8192

  // workspace carve-out (all f32-sized slots, 16B aligned by construction)
  float* ws = (float*)d_ws;
  size_t off = 0;
  auto carve = [&](size_t n) { float* p = ws + off; off += n; return p; };
  float*    x    = carve((size_t)N * F);
  float*    ft   = carve((size_t)N * F);
  float*    res  = carve((size_t)N * F);
  float*    agg  = carve((size_t)N * F);
  float*    el   = carve((size_t)N * H);
  float*    er   = carve((size_t)N * H);
  unsigned* nmax = (unsigned*)carve((size_t)N * H);
  float*    nsum = carve((size_t)N * H);
  float*    ebuf = carve((size_t)E * H);
  float*    abuf = carve((size_t)E * H);

  const dim3 blk(256);
  auto gemm = [&](const float* A, const float* B, float* Cm, int M, int Nn,
                  int K) {
    dim3 grid(Nn / 128, M / 128);
    gemm_bf16wmma_kernel<<<grid, blk, 0, stream>>>(A, B, Cm, M, Nn, K);
  };
  auto run_gat = [&](const float* ftL, const float* alL, const float* arL,
                     int Hh, float* aggL) {
    int NH = N * Hh, EH = E * Hh, HD = Hh * D;
    scores_kernel<<<(NH * 32 + 255) / 256, blk, 0, stream>>>(ftL, alL, arL, el,
                                                             er, NH, Hh, D);
    zero_kernel<<<(NH + 255) / 256, blk, 0, stream>>>((float*)nmax, (size_t)NH);
    zero_kernel<<<(NH + 255) / 256, blk, 0, stream>>>(nsum, (size_t)NH);
    edge_max_kernel<<<(EH + 255) / 256, blk, 0, stream>>>(el, er, src, dst,
                                                          ebuf, nmax, EH, Hh);
    edge_exp_kernel<<<(EH + 255) / 256, blk, 0, stream>>>(ebuf, nmax, nsum, dst,
                                                          EH, Hh);
    edge_norm_kernel<<<(EH + 255) / 256, blk, 0, stream>>>(ebuf, nsum, dst,
                                                           abuf, EH, Hh);
    size_t aggN = (size_t)N * HD;
    zero_kernel<<<(int)((aggN + 255) / 256), blk, 0, stream>>>(aggL, aggN);
    aggregate_kernel<<<E, blk, 0, stream>>>(ftL, abuf, src, dst, aggL, HD, 9);
  };

  // ---- layer 1: ft = feat@W1, res = feat@resW1 (Linear residual)
  gemm(feat, W1, ft, N, F, C);
  gemm(feat, resW1, res, N, F, C);
  run_gat(ft, al1, ar1, H, agg);
  add_relu_kernel<<<(int)(((size_t)N * F + 255) / 256), blk, 0, stream>>>(
      agg, res, x, (size_t)N * F);

  // ---- layer 2: identity residual
  gemm(x, W2, ft, N, F, F);
  run_gat(ft, al2, ar2, H, agg);
  add_relu_kernel<<<(int)(((size_t)N * F + 255) / 256), blk, 0, stream>>>(
      agg, x, x, (size_t)N * F);

  // ---- layer 3: identity residual
  gemm(x, W3, ft, N, F, F);
  run_gat(ft, al3, ar3, H, agg);
  add_relu_kernel<<<(int)(((size_t)N * F + 255) / 256), blk, 0, stream>>>(
      agg, x, x, (size_t)N * F);

  // ---- layer 4: 1 head, Linear residual
  gemm(x, W4, ft, N, D, F);
  gemm(x, resW4, res, N, D, F);
  run_gat(ft, al4, ar4, 1, agg);

  int ND = N * D;
  final_kernel<<<(2 * ND + E + 255) / 256, blk, 0, stream>>>(
      agg, res, abuf, (float*)d_out, ND, E);
}
